// Packetbnn_71339406787195
// MI455X (gfx1250) — 8-bit matrix operand order repaired and checked by reading, compiled, not run on hardware
//
#include <hip/hip_runtime.h>
#include <stdint.h>

typedef __attribute__((ext_vector_type(8))) int   v8i;
typedef __attribute__((ext_vector_type(4))) int   v4i;
typedef __attribute__((ext_vector_type(2))) float v2f;

#define FDIM 126
#define FPAD 128
#define GN_EPS 1e-5f

__device__ __forceinline__ int isgn(float v) { return (v > 0.f) - (v < 0.f); }

__device__ __forceinline__ float rsum16(float v) {
  v += __shfl_xor(v, 1, 32);
  v += __shfl_xor(v, 2, 32);
  v += __shfl_xor(v, 4, 32);
  v += __shfl_xor(v, 8, 32);
  return v;
}

__global__ __launch_bounds__(256) void pack_w_kernel(const float* __restrict__ conv_w,
                                                     int8_t* __restrict__ wb) {
  int idx = blockIdx.x * 256 + threadIdx.x;
  int c = idx >> 7, f = idx & 127;
  float v = (c < FDIM && f < FDIM) ? conv_w[c * FDIM + f] : 0.f;
  wb[idx] = (int8_t)isgn(v);
}

__global__ __launch_bounds__(256, 1) void bnn_fused_kernel(
    const float* __restrict__ x, const int8_t* __restrict__ wb,
    const float* __restrict__ gn1w, const float* __restrict__ gn1b,
    const float* __restrict__ gn2w, const float* __restrict__ gn2b,
    const float* __restrict__ linw, const float* __restrict__ linb,
    const float* __restrict__ gn3w, const float* __restrict__ gn3b,
    float* __restrict__ out) {
  __shared__ int8_t wlds[FPAD * FPAD];
  __shared__ float sg1w[FPAD], sg1b[FPAD], sg2w[FPAD], sg2b[FPAD];
  __shared__ float slw0[FPAD], slw1[FPAD];

  const int tid  = threadIdx.x;
  const int wave = tid >> 5;
  const int lane = tid & 31;
  const int half = lane >> 4;
  const int l16  = lane & 15;

  {
    const v4i* src = (const v4i*)wb;
    v4i* dst = (v4i*)wlds;
#pragma unroll
    for (int i = 0; i < 4; ++i) dst[i * 256 + tid] = src[i * 256 + tid];
  }
  if (tid < FPAD) {
    bool v = tid < FDIM;
    sg1w[tid] = v ? gn1w[tid] : 1.f;
    sg1b[tid] = v ? gn1b[tid] : 0.f;
    sg2w[tid] = v ? gn2w[tid] : 1.f;
    sg2b[tid] = v ? gn2b[tid] : 0.f;
    float a0 = v ? linw[tid] : 0.f;
    float a1 = v ? linw[FDIM + tid] : 0.f;
    slw0[tid] = (float)isgn(a0);
    slw1[tid] = (float)isgn(a1);
  }

  const int rowbase = blockIdx.x * 128 + wave * 16;
  const float* xrow = x + (size_t)(rowbase + l16) * FDIM;
  __builtin_prefetch(xrow, 0, 1);

  v8i afrag[2];
#pragma unroll
  for (int kt = 0; kt < 2; ++kt) {
#pragma unroll
    for (int g = 0; g < 8; ++g) {
      int k0 = kt * 64 + (g >> 2) * 32 + half * 16 + (g & 3) * 4;
      v2f lo = *(const v2f*)(xrow + k0);
      v2f hi;
      if (k0 + 2 < FDIM) hi = *(const v2f*)(xrow + k0 + 2);
      else { hi.x = 0.f; hi.y = 0.f; }
      uint32_t pk = (uint32_t)(uint8_t)(int8_t)isgn(lo.x)
                  | ((uint32_t)(uint8_t)(int8_t)isgn(lo.y) << 8)
                  | ((uint32_t)(uint8_t)(int8_t)isgn(hi.x) << 16)
                  | ((uint32_t)(uint8_t)(int8_t)isgn(hi.y) << 24);
      afrag[kt][g] = (int)pk;
    }
  }

  __syncthreads();

  v8i acc[8];
#pragma unroll
  for (int nt = 0; nt < 8; ++nt) {
    v8i z = {0, 0, 0, 0, 0, 0, 0, 0};
    acc[nt] = z;
  }
#pragma unroll
  for (int kt = 0; kt < 2; ++kt) {
    v8i a = afrag[kt];
#pragma unroll
    for (int nt = 0; nt < 8; ++nt) {
      int c = nt * 16 + l16;
      int base = c * FPAD + kt * 64 + half * 16;
      v4i b0 = *(const v4i*)&wlds[base];
      v4i b1 = *(const v4i*)&wlds[base + 32];
      v8i b;
      b[0] = b0.x; b[1] = b0.y; b[2] = b0.z; b[3] = b0.w;
      b[4] = b1.x; b[5] = b1.y; b[6] = b1.z; b[7] = b1.w;
      acc[nt] = __builtin_amdgcn_wmma_i32_16x16x64_iu8(
          true, a, true, b, acc[nt], false, false);
    }
  }

  float hv[8][8];
  float s1[8], s2[8];
#pragma unroll
  for (int j = 0; j < 8; ++j) { s1[j] = 0.f; s2[j] = 0.f; }
#pragma unroll
  for (int nt = 0; nt < 8; ++nt) {
    bool valid = (nt * 16 + l16) < FDIM;
#pragma unroll
    for (int j = 0; j < 8; ++j) {
      float h = (float)acc[nt][j];
      hv[nt][j] = h;
      if (valid) { s1[j] += h; s2[j] += h * h; }
    }
  }

  const float invF = 1.f / (float)FDIM;
  float mean[8], rstd[8];
#pragma unroll
  for (int j = 0; j < 8; ++j) {
    float S1 = rsum16(s1[j]);
    float S2 = rsum16(s2[j]);
    float mu = S1 * invF;
    float var = S2 * invF - mu * mu;
    mean[j] = mu;
    rstd[j] = rsqrtf(var + GN_EPS);
  }

#pragma unroll
  for (int j = 0; j < 8; ++j) { s1[j] = 0.f; s2[j] = 0.f; }
#pragma unroll
  for (int nt = 0; nt < 8; ++nt) {
    int n = nt * 16 + l16;
    bool valid = n < FDIM;
    float g = sg1w[n], b = sg1b[n];
#pragma unroll
    for (int j = 0; j < 8; ++j) {
      float h = (hv[nt][j] - mean[j]) * rstd[j] * g + b;
      h = fminf(1.f, fmaxf(-1.f, h));
      hv[nt][j] = h;
      if (valid) { s1[j] += h; s2[j] += h * h; }
    }
  }
#pragma unroll
  for (int j = 0; j < 8; ++j) {
    float S1 = rsum16(s1[j]);
    float S2 = rsum16(s2[j]);
    float mu = S1 * invF;
    float var = S2 * invF - mu * mu;
    mean[j] = mu;
    rstd[j] = rsqrtf(var + GN_EPS);
  }

  float p0[8], p1[8];
#pragma unroll
  for (int j = 0; j < 8; ++j) { p0[j] = 0.f; p1[j] = 0.f; }
#pragma unroll
  for (int nt = 0; nt < 8; ++nt) {
    int n = nt * 16 + l16;
    bool valid = n < FDIM;
    float g = sg2w[n], b = sg2b[n];
    float w0 = slw0[n], w1 = slw1[n];
#pragma unroll
    for (int j = 0; j < 8; ++j) {
      float h = (hv[nt][j] - mean[j]) * rstd[j] * g + b;
      h = fminf(1.f, fmaxf(-1.f, h));
      float hb = (float)isgn(h);
      if (valid) { p0[j] += hb * w0; p1[j] += hb * w1; }
    }
  }

  float lb0 = linb[0], lb1 = linb[1];
  float g30 = gn3w[0], g31 = gn3w[1];
  float b30 = gn3b[0], b31 = gn3b[1];

#pragma unroll
  for (int j = 0; j < 8; ++j) {
    float O0 = rsum16(p0[j]) + lb0;
    float O1 = rsum16(p1[j]) + lb1;
    float mu = 0.5f * (O0 + O1);
    float d0 = O0 - mu, d1 = O1 - mu;
    float var = 0.5f * (d0 * d0 + d1 * d1);
    float inv = rsqrtf(var + GN_EPS);
    float r0 = d0 * inv * g30 + b30;
    float r1 = d1 * inv * g31 + b31;
    if (l16 == 0) {
      int row = rowbase + half * 8 + j;
      v2f o; o.x = r0; o.y = r1;
      *(v2f*)(out + (size_t)row * 2) = o;
    }
  }
}

extern "C" void kernel_launch(void* const* d_in, const int* in_sizes, int n_in,
                              void* d_out, int out_size, void* d_ws, size_t ws_size,
                              hipStream_t stream) {
  const float* x      = (const float*)d_in[0];
  const float* conv_w = (const float*)d_in[1];
  const float* gn1w   = (const float*)d_in[2];
  const float* gn1b   = (const float*)d_in[3];
  const float* gn2w   = (const float*)d_in[4];
  const float* gn2b   = (const float*)d_in[5];
  const float* linw   = (const float*)d_in[6];
  const float* linb   = (const float*)d_in[7];
  const float* gn3w   = (const float*)d_in[8];
  const float* gn3b   = (const float*)d_in[9];
  float* out = (float*)d_out;
  int8_t* wb = (int8_t*)d_ws;

  pack_w_kernel<<<64, 256, 0, stream>>>(conv_w, wb);

  int B = in_sizes[0] / FDIM;
  int nblocks = B / 128;
  bnn_fused_kernel<<<nblocks, 256, 0, stream>>>(
      x, wb, gn1w, gn1b, gn2w, gn2b, linw, linb, gn3w, gn3b, out);
}
